// MinkUNet_30081950941516
// MI455X (gfx1250) — matrix operand order repaired and checked by reading, not compiled, not run on hardware
//
#include <hip/hip_runtime.h>
#include <hip/hip_bf16.h>
#include <cstdint>

// ---------------------------------------------------------------------------
// CDNA5 / gfx1250 sparse-conv stack via v_wmma_f32_16x16x32_bf16 with
// LDS-staged weight tiles (double-buffered) and 2 row-tiles per wave.
// ---------------------------------------------------------------------------

typedef __attribute__((ext_vector_type(16))) __bf16 v16bf;
typedef __attribute__((ext_vector_type(8)))  float  v8f;

// ---------------------------------------------------------------------------
// Pack fp32 weights [K, Cin, Cout] into bf16 WMMA-B tiles.
// Tile = 32(K-dim) x 16(N-dim) bf16 = 512 elems; lane L's fragment is the 16
// consecutive bf16 at tile + L*16:
//   lane 0-15  : column = lane,    K = elem      (K 0..15)
//   lane 16-31 : column = lane-16, K = 16 + elem (K 16..31)
// Tile order: ((k * cinB + cb) * coutB + ob)
// ---------------------------------------------------------------------------
__global__ void pack_weights(const float* __restrict__ W, __bf16* __restrict__ out,
                             int K, int Cin, int Cout) {
  const int cinB  = (Cin + 31) / 32;
  const int coutB = Cout / 16;
  const int total = K * cinB * coutB * 512;
  int tid = blockIdx.x * blockDim.x + threadIdx.x;
  if (tid >= total) return;
  int e    = tid & 15;
  int lane = (tid >> 4) & 31;
  int tile = tid >> 9;
  int ob = tile % coutB;
  int cb = (tile / coutB) % cinB;
  int k  = tile / (coutB * cinB);
  int cin  = cb * 32 + (lane >= 16 ? 16 : 0) + e;
  int cout = ob * 16 + (lane & 15);
  float v = (cin < Cin) ? W[((size_t)k * Cin + cin) * Cout + cout] : 0.0f;
  out[tid] = (__bf16)v;
}

// ---------------------------------------------------------------------------
// Gather-GEMM-scatter sparse conv with fused BN (+ optional skip add, ReLU).
// 256 threads = 8 waves; each wave computes 2 x (16-row x COUT) tiles.
// Per k: the block stages all CINB*COUTB B-tiles for that kernel offset into
// LDS once (global->reg->LDS double buffer), then every wave reads fragments
// from LDS and issues 2 WMMAs per fragment (one per row tile).
// A fragment (16x32 bf16), built from gathered rows:
//   lane 0-15  : row = lane,    elems -> K {0..15}
//   lane 16-31 : row = lane-16, elems -> K {16..31}
// ---------------------------------------------------------------------------
template<int CIN, int COUT, bool IDENT, bool ADDSKIP, bool RELU>
__global__ __launch_bounds__(256) void gconv(
    const float* __restrict__ X, const __bf16* __restrict__ Wp,
    const int* __restrict__ nbr, const float* __restrict__ nmask,
    const float* __restrict__ bn, const float* __restrict__ skip,
    float* __restrict__ Y, int N, int K) {
  constexpr int CINB    = (CIN + 31) / 32;
  constexpr int COUTB   = COUT / 16;
  constexpr int KTILES  = CINB * COUTB;          // B tiles per kernel offset
  constexpr int KELEMS  = KTILES * 512;          // bf16 per k slice
  constexpr int KCHUNKS = (KELEMS * 2) / 16;     // float4 chunks per k slice
  constexpr int NCHUNK  = (KCHUNKS + 255) / 256; // chunks per thread

  __shared__ __align__(32) __bf16 smem[2 * KELEMS];

  const int tid     = threadIdx.x;
  const int lane    = tid & 31;
  const int wave    = tid >> 5;
  const int tileRow = (blockIdx.x * 8 + wave) * 32;   // 2 x 16 rows per wave
  const bool hiHalf = (lane >= 16);

  const int r0 = tileRow + (lane & 15);
  const int r1 = r0 + 16;
  const int rc0 = (r0 < N) ? r0 : (N - 1);            // clamped: loads always legal
  const int rc1 = (r1 < N) ? r1 : (N - 1);
  const float inb0 = (r0 < N) ? 1.0f : 0.0f;
  const float inb1 = (r1 < N) ? 1.0f : 0.0f;

  v8f acc0[COUTB] = {};
  v8f acc1[COUTB] = {};

  // Branchless A-fragment build from a gathered, masked row.
  auto buildA = [&](int idx, float m, int cb) -> v16bf {
    v16bf A;
    if constexpr (CIN == 3) {
      const float* rp = X + (size_t)idx * 3;
      float s = hiHalf ? 0.0f : m;                     // only lane<16 K=0..2 valid
      A[0] = (__bf16)(rp[0] * s);
      A[1] = (__bf16)(rp[1] * s);
      A[2] = (__bf16)(rp[2] * s);
      #pragma unroll
      for (int e = 3; e < 16; ++e) A[e] = (__bf16)0.0f;
    } else {
      const float4* p =
          (const float4*)(X + (size_t)idx * CIN + cb * 32 + (hiHalf ? 16 : 0));
      float4 a0 = p[0], a1 = p[1], b0 = p[2], b1 = p[3];
      A[0]  = (__bf16)(a0.x * m); A[1]  = (__bf16)(a0.y * m);
      A[2]  = (__bf16)(a0.z * m); A[3]  = (__bf16)(a0.w * m);
      A[4]  = (__bf16)(a1.x * m); A[5]  = (__bf16)(a1.y * m);
      A[6]  = (__bf16)(a1.z * m); A[7]  = (__bf16)(a1.w * m);
      A[8]  = (__bf16)(b0.x * m); A[9]  = (__bf16)(b0.y * m);
      A[10] = (__bf16)(b0.z * m); A[11] = (__bf16)(b0.w * m);
      A[12] = (__bf16)(b1.x * m); A[13] = (__bf16)(b1.y * m);
      A[14] = (__bf16)(b1.z * m); A[15] = (__bf16)(b1.w * m);
    }
    return A;
  };

  const int kEnd = IDENT ? 1 : K;
  float4* smemf4 = (float4*)smem;

  // Prefetch k=0 weight slice into registers.
  float4 sreg[NCHUNK];
  {
    const float4* src = (const float4*)Wp;
    #pragma unroll
    for (int i = 0; i < NCHUNK; ++i) {
      int gi = tid + i * 256;
      if (gi < KCHUNKS) sreg[i] = src[gi];
    }
  }

  for (int k = 0; k < kEnd; ++k) {
    const int cur = k & 1;
    // Commit staged registers to this k's LDS buffer.
    #pragma unroll
    for (int i = 0; i < NCHUNK; ++i) {
      int gi = tid + i * 256;
      if (gi < KCHUNKS) smemf4[cur * KCHUNKS + gi] = sreg[i];
    }
    __syncthreads();
    // Issue next slice's global loads (overlaps with this k's compute).
    if (k + 1 < kEnd) {
      const float4* src = (const float4*)(Wp + (size_t)(k + 1) * KELEMS);
      #pragma unroll
      for (int i = 0; i < NCHUNK; ++i) {
        int gi = tid + i * 256;
        if (gi < KCHUNKS) sreg[i] = src[gi];
      }
    }

    // Gather neighbor index + mask (branchless, clamped addresses).
    int idx0, idx1; float m0, m1;
    if constexpr (IDENT) {
      idx0 = rc0; m0 = inb0;
      idx1 = rc1; m1 = inb1;
    } else {
      const size_t kb = (size_t)k * N;
      idx0 = nbr[kb + rc0];
      idx1 = nbr[kb + rc1];
      m0 = (nmask[kb + rc0] > 0.1f) ? inb0 : 0.0f;
      m1 = (nmask[kb + rc1] > 0.1f) ? inb1 : 0.0f;
    }

    const __bf16* wk = smem + cur * KELEMS;
    #pragma unroll
    for (int cb = 0; cb < CINB; ++cb) {
      v16bf A0 = buildA(idx0, m0, cb);
      v16bf A1 = buildA(idx1, m1, cb);
      #pragma unroll
      for (int ob = 0; ob < COUTB; ++ob) {
        v16bf B = *(const v16bf*)(wk + (cb * COUTB + ob) * 512 + lane * 16);
        acc0[ob] = __builtin_amdgcn_wmma_f32_16x16x32_bf16(
            false, A0, false, B, (short)0, acc0[ob], false, false);
        acc1[ob] = __builtin_amdgcn_wmma_f32_16x16x32_bf16(
            false, A1, false, B, (short)0, acc1[ob], false, false);
      }
    }
  }

  // Epilogue: BN per channel, optional skip add + ReLU.
  // C/D layout: elem j of lane L -> row = tileBase + j + (L>=16 ? 8:0), col = L&15.
  const int col     = lane & 15;
  const int rowOff0 = tileRow + (hiHalf ? 8 : 0);
  #pragma unroll
  for (int ob = 0; ob < COUTB; ++ob) {
    int ch = ob * 16 + col;
    float g   = bn[0 * COUT + ch];
    float be  = bn[1 * COUT + ch];
    float mu  = bn[2 * COUT + ch];
    float var = bn[3 * COUT + ch];
    float scale = g * rsqrtf(var + 1e-5f);
    float shift = be - mu * scale;
    #pragma unroll
    for (int j = 0; j < 8; ++j) {
      int row = rowOff0 + j;
      if (row < N) {
        float v = acc0[ob][j] * scale + shift;
        if constexpr (ADDSKIP) v += skip[(size_t)row * COUT + ch];
        if constexpr (RELU)    v = fmaxf(v, 0.0f);
        Y[(size_t)row * COUT + ch] = v;
      }
      int row1 = row + 16;
      if (row1 < N) {
        float v = acc1[ob][j] * scale + shift;
        if constexpr (ADDSKIP) v += skip[(size_t)row1 * COUT + ch];
        if constexpr (RELU)    v = fmaxf(v, 0.0f);
        Y[(size_t)row1 * COUT + ch] = v;
      }
    }
  }
}

// ---------------------------------------------------------------------------
// Pruning classifier: gathered 64->1 conv + bias, sigmoid.  cls>0.5 <=> logit>0.
// ---------------------------------------------------------------------------
__global__ void prune_kernel(const float* __restrict__ x2, const int* __restrict__ nbr,
                             const float* __restrict__ nmask, const float* __restrict__ wp,
                             const float* __restrict__ bp, float* __restrict__ cls_out,
                             float* __restrict__ keep_out, int N, int K) {
  int i = blockIdx.x * blockDim.x + threadIdx.x;
  if (i >= N) return;
  float acc = bp[0];
  for (int k = 0; k < K; ++k) {
    int idx = nbr[(size_t)k * N + i];
    bool m  = nmask[(size_t)k * N + i] > 0.1f;
    if (m) {
      const float4* row = (const float4*)(x2 + (size_t)idx * 64);
      const float4* w   = (const float4*)(wp + (size_t)k * 64);
      float s = 0.0f;
      #pragma unroll
      for (int c = 0; c < 16; ++c) {
        float4 rv = row[c], wv = w[c];
        s += rv.x * wv.x + rv.y * wv.y + rv.z * wv.z + rv.w * wv.w;
      }
      acc += s;
    }
  }
  cls_out[i]  = 1.0f / (1.0f + __expf(-acc));
  keep_out[i] = (acc > 0.0f) ? 1.0f : 0.0f;
}

__global__ void mask_out_kernel(const float* __restrict__ x2, const float* __restrict__ keep,
                                float* __restrict__ out, size_t total) {
  size_t t = blockIdx.x * (size_t)blockDim.x + threadIdx.x;
  if (t < total) out[t] = x2[t] * keep[t >> 6];
}

// ---------------------------------------------------------------------------
// Host-side orchestration.
// ---------------------------------------------------------------------------
static inline size_t align_up(size_t x, size_t a) { return (x + a - 1) & ~(a - 1); }

extern "C" void kernel_launch(void* const* d_in, const int* in_sizes, int n_in,
                              void* d_out, int out_size, void* d_ws, size_t ws_size,
                              hipStream_t stream) {
  const float* feats   = (const float*)d_in[0];   // [N,3]
  const int*   nbr     = (const int*)  d_in[1];   // [K,N]
  const float* nmask   = (const float*)d_in[2];   // [K,N]
  const float* w_stem  = (const float*)d_in[3];   // [K,3,32]
  const float* bn_stem = (const float*)d_in[4];
  const float* w_r1a   = (const float*)d_in[5];   // [K,32,64]
  const float* bn_r1a  = (const float*)d_in[6];
  const float* w_r1b   = (const float*)d_in[7];   // [K,64,64]
  const float* bn_r1b  = (const float*)d_in[8];
  const float* w_r1d   = (const float*)d_in[9];   // [32,64]
  const float* bn_r1d  = (const float*)d_in[10];
  const float* w_r2a   = (const float*)d_in[11];  // [K,64,64]
  const float* bn_r2a  = (const float*)d_in[12];
  const float* w_r2b   = (const float*)d_in[13];  // [K,64,64]
  const float* bn_r2b  = (const float*)d_in[14];
  const float* w_prune = (const float*)d_in[15];  // [K,64,1]
  const float* b_prune = (const float*)d_in[16];  // [1]

  const int N = in_sizes[0] / 3;
  const int K = in_sizes[1] / N;     // 27

  // ---- workspace carve-out -------------------------------------------------
  char* ws = (char*)d_ws;
  size_t off = 0;
  auto carve = [&](size_t bytes) -> void* {
    void* p = ws + off;
    off = align_up(off + bytes, 256);
    return p;
  };
  const size_t t_stem = (size_t)K * 1 * 2 * 512;   // cinB=1, coutB=2
  const size_t t_c64  = (size_t)K * 1 * 4 * 512;   // 32->64
  const size_t t_d    = (size_t)1 * 1 * 4 * 512;   // 1x1 32->64
  const size_t t_c128 = (size_t)K * 2 * 4 * 512;   // 64->64
  __bf16* wp_stem = (__bf16*)carve(t_stem * 2);
  __bf16* wp_r1a  = (__bf16*)carve(t_c64  * 2);
  __bf16* wp_r1d  = (__bf16*)carve(t_d    * 2);
  __bf16* wp_r1b  = (__bf16*)carve(t_c128 * 2);
  __bf16* wp_r2a  = (__bf16*)carve(t_c128 * 2);
  __bf16* wp_r2b  = (__bf16*)carve(t_c128 * 2);
  float* keepbuf = (float*)carve((size_t)N * sizeof(float));
  float* x0      = (float*)carve((size_t)N * 32 * sizeof(float));  // stem out
  float* bufB    = (float*)carve((size_t)N * 64 * sizeof(float));  // h1 / h3
  float* bufC    = (float*)carve((size_t)N * 64 * sizeof(float));  // skip / x2
  float* bufD    = (float*)carve((size_t)N * 64 * sizeof(float));  // x1
  (void)ws_size; (void)n_in; (void)out_size;

  // ---- pack weights --------------------------------------------------------
  auto packGrid = [](size_t total) { return dim3((unsigned)((total + 255) / 256)); };
  pack_weights<<<packGrid(t_stem), 256, 0, stream>>>(w_stem, wp_stem, K, 3, 32);
  pack_weights<<<packGrid(t_c64 ), 256, 0, stream>>>(w_r1a,  wp_r1a,  K, 32, 64);
  pack_weights<<<packGrid(t_d   ), 256, 0, stream>>>(w_r1d,  wp_r1d,  1, 32, 64);
  pack_weights<<<packGrid(t_c128), 256, 0, stream>>>(w_r1b,  wp_r1b,  K, 64, 64);
  pack_weights<<<packGrid(t_c128), 256, 0, stream>>>(w_r2a,  wp_r2a,  K, 64, 64);
  pack_weights<<<packGrid(t_c128), 256, 0, stream>>>(w_r2b,  wp_r2b,  K, 64, 64);

  // ---- network -------------------------------------------------------------
  const dim3 grid((N + 255) / 256);   // 8 waves x 32 rows per block
  const dim3 blk(256);

  // stem: conv3(3->32) + BN + ReLU
  gconv<3, 32, false, false, true><<<grid, blk, 0, stream>>>(
      feats, wp_stem, nbr, nmask, bn_stem, nullptr, x0, N, K);
  // skip projection: x0 @ Wd (1x1) + BN (no relu)
  gconv<32, 64, true, false, false><<<grid, blk, 0, stream>>>(
      x0, wp_r1d, nullptr, nullptr, bn_r1d, nullptr, bufC, N, K);
  // r1a: conv3(32->64) + BN + ReLU
  gconv<32, 64, false, false, true><<<grid, blk, 0, stream>>>(
      x0, wp_r1a, nbr, nmask, bn_r1a, nullptr, bufB, N, K);
  // r1b: conv3(64->64) + BN, + skip, ReLU  -> x1
  gconv<64, 64, false, true, true><<<grid, blk, 0, stream>>>(
      bufB, wp_r1b, nbr, nmask, bn_r1b, bufC, bufD, N, K);
  // r2a: conv3(64->64) + BN + ReLU
  gconv<64, 64, false, false, true><<<grid, blk, 0, stream>>>(
      bufD, wp_r2a, nbr, nmask, bn_r2a, nullptr, bufB, N, K);
  // r2b: conv3(64->64) + BN, + x1, ReLU -> x2
  gconv<64, 64, false, true, true><<<grid, blk, 0, stream>>>(
      bufB, wp_r2b, nbr, nmask, bn_r2b, bufD, bufC, N, K);

  // ---- prune + outputs -----------------------------------------------------
  float* out  = (float*)d_out;
  float* cls  = out + (size_t)N * 64;
  prune_kernel<<<dim3((N + 255) / 256), 256, 0, stream>>>(
      bufC, nbr, nmask, w_prune, b_prune, cls, keepbuf, N, K);
  const size_t total = (size_t)N * 64;
  mask_out_kernel<<<dim3((unsigned)((total + 255) / 256)), 256, 0, stream>>>(
      bufC, keepbuf, out, total);
}
